// TokenLearner_17171279250009
// MI455X (gfx1250) — matrix operand order repaired and checked by reading, not compiled, not run on hardware
//
#include <hip/hip_runtime.h>
#include <stdint.h>

// ---------- types ----------
typedef __attribute__((ext_vector_type(16))) __bf16 v16bf;
typedef __attribute__((ext_vector_type(8)))  __bf16 v8bf;
typedef __attribute__((ext_vector_type(8)))  float  v8f;
typedef int v4i_vs __attribute__((vector_size(16)));   // matches builtin param type

#define B_   32
#define N_   4096
#define C_   768
#define K_   8
#define MTOT (B_ * N_)      // 131072 rows
#define MT   64             // rows per workgroup
#define LDP  80             // padded LDS row length (bf16 elems), 160B = 16B aligned
#define NTILES (C_ / 64)    // 12

// ---------- CDNA5 async global->LDS copy (ASYNCcnt path), with safe fallback ----
#if __has_builtin(__builtin_amdgcn_global_load_async_to_lds_b128)
#define TL_ASYNC 1
static __device__ __forceinline__ void tl_async_cp16(const __bf16* g, __bf16* l) {
    __builtin_amdgcn_global_load_async_to_lds_b128(
        (__attribute__((address_space(1))) v4i_vs*)(uintptr_t)g,
        (__attribute__((address_space(3))) v4i_vs*)(uint32_t)(uintptr_t)l,
        /*offset=*/0, /*cpol=*/0);
}
static __device__ __forceinline__ void tl_wait_async_le2() {
#if __has_builtin(__builtin_amdgcn_s_wait_asynccnt)
    __builtin_amdgcn_s_wait_asynccnt(2);
#else
    asm volatile("s_wait_asynccnt 0x2" ::: "memory");
#endif
}
#else
#define TL_ASYNC 0
static __device__ __forceinline__ void tl_async_cp16(const __bf16* g, __bf16* l) {
    *(v8bf*)l = *(const v8bf*)g;   // synchronous fallback
}
static __device__ __forceinline__ void tl_wait_async_le2() {}
#endif

static __device__ __forceinline__ v8f zero8() {
    v8f z;
#pragma unroll
    for (int i = 0; i < 8; i++) z[i] = 0.f;
    return z;
}

static __device__ __forceinline__ v16bf cat16(v8bf lo, v8bf hi) {
    v16bf r;
#pragma unroll
    for (int i = 0; i < 8; i++) { r[i] = lo[i]; r[i + 8] = hi[i]; }
    return r;
}

// 32B contiguous (two 16B LDS/global loads)
static __device__ __forceinline__ v16bf load16c(const __bf16* p) {
    return cat16(*(const v8bf*)p, *(const v8bf*)(p + 8));
}

// A-fragment: two 16B chunks at o0 and o1 (callers use o1 = o0 + 8: contiguous K 16h..16h+15)
static __device__ __forceinline__ v16bf loadA2(const __bf16* row, int o0, int o1) {
    return cat16(*(const v8bf*)(row + o0), *(const v8bf*)(row + o1));
}

static __device__ __forceinline__ float gelu_exact(float v) {
    return 0.5f * v * (1.f + erff(v * 0.70710678118654752440f));
}

// ---------- kernel 0: weight prep (f32 -> transposed bf16) ----------
__global__ void tl_prep(const float* __restrict__ W1, const float* __restrict__ W2,
                        __bf16* __restrict__ W1T, __bf16* __restrict__ W2T) {
    int i = blockIdx.x * 256 + threadIdx.x;
    if (i < C_ * C_) {
        int n = i / C_, k = i % C_;
        W1T[i] = (__bf16)W1[k * C_ + n];             // W1T[n][k] = W1[k][n]
    }
    if (i < 16 * C_) {
        int n = i / C_, k = i % C_;
        W2T[i] = (n < K_) ? (__bf16)W2[k * K_ + n] : (__bf16)0.f;  // pad cols 8..15
    }
}

// ---------- kernel 1: fused LN + GEMM1(+b1,GELU) + GEMM2 -> logits ----------
__global__ void __launch_bounds__(256)
tl_fused(const float* __restrict__ x, const float* __restrict__ gam,
         const float* __restrict__ bet, const __bf16* __restrict__ W1T,
         const float* __restrict__ b1, const __bf16* __restrict__ W2T,
         float* __restrict__ logits) {
    __shared__ __bf16 Abuf[MT][LDP];        // normalized x chunk [m][k], k-chunk of 64
    __shared__ __bf16 Bbuf[2][MT][LDP];     // W1T chunk [n][k], double-buffered (async)
    __shared__ __bf16 Hbuf[MT][LDP];        // GELU(h) tile [m][n], n-tile of 64
    __shared__ float  row_mu[MT], row_rs[MT];

    const int tid  = threadIdx.x;
    const int lane = tid & 31;
    const int wave = tid >> 5;          // 0..7
    const int rowblk = wave & 3;        // 16-row block
    const int colgrp = wave >> 2;       // 32-col group
    const int lh = lane & 15;
    const int hb = lane >> 4;           // half-wave index
    const int kb = hb * 8;              // C-frag row offset

    const int sr  = tid & 63;           // staging row (A) / col (B)
    const int seg = tid >> 6;           // 0..3 -> 16 k elems each

    const size_t m0 = (size_t)blockIdx.x * MT;

    // ---- LayerNorm statistics: one wave per 8 rows, wave32 shuffle reduction ----
    for (int r = wave * 8; r < wave * 8 + 8; ++r) {
        const float* xr = x + (m0 + r) * C_;
        float s = 0.f, sq = 0.f;
#pragma unroll
        for (int i = 0; i < 6; i++) {
            float4 v = *(const float4*)(xr + i * 128 + lane * 4);
            s  += v.x + v.y + v.z + v.w;
            sq += v.x * v.x + v.y * v.y + v.z * v.z + v.w * v.w;
        }
#pragma unroll
        for (int off = 16; off; off >>= 1) {
            s  += __shfl_xor(s,  off, 32);
            sq += __shfl_xor(sq, off, 32);
        }
        if (lane == 0) {
            float mu  = s * (1.f / C_);
            float var = sq * (1.f / C_) - mu * mu;
            row_mu[r] = mu;
            row_rs[r] = rsqrtf(var + 1e-5f);
        }
    }

    v8f lacc = zero8();                 // persistent logits accumulator (waves 0..3)

    // Prime the async pipeline: B tile (ct=0, kt=0) -> Bbuf[0]  (2x b128 per thread)
    {
        const __bf16* gp = W1T + (size_t)sr * C_ + seg * 16;
        tl_async_cp16(gp,     &Bbuf[0][sr][seg * 16]);
        tl_async_cp16(gp + 8, &Bbuf[0][sr][seg * 16 + 8]);
    }
    __syncthreads();                    // also covers row_mu/row_rs

    for (int ct = 0; ct < NTILES; ++ct) {           // 12 output-column tiles
        v8f acc0 = zero8(), acc1 = zero8();

        for (int kt = 0; kt < NTILES; ++kt) {       // 12 k-chunks of 64
            const int buf = kt & 1;
            __syncthreads();            // all waves done reading Abuf / Bbuf[buf^1]

            // ---- issue async prefetch of the NEXT B tile into the other buffer ----
            {
                int nct = ct, nkt = kt + 1;
                if (nkt == NTILES) { nkt = 0; if (ct + 1 < NTILES) nct = ct + 1; }
                const __bf16* gp =
                    W1T + (size_t)(nct * 64 + sr) * C_ + nkt * 64 + seg * 16;
                tl_async_cp16(gp,     &Bbuf[buf ^ 1][sr][seg * 16]);
                tl_async_cp16(gp + 8, &Bbuf[buf ^ 1][sr][seg * 16 + 8]);
            }

            // ---- stage A into LDS (normalize on the fly), overlaps async copy ----
            {
                const int kg  = kt * 64 + seg * 16;
                const float* xp = x + (m0 + sr) * C_ + kg;
                __builtin_prefetch(xp + 64, 0, 1);  // next k-chunk -> global_prefetch_b8
                const float mu = row_mu[sr], rs = row_rs[sr];
                float vals[16];
#pragma unroll
                for (int i = 0; i < 4; i++) {
                    float4 v = *(const float4*)(xp + i * 4);
                    vals[i * 4 + 0] = v.x; vals[i * 4 + 1] = v.y;
                    vals[i * 4 + 2] = v.z; vals[i * 4 + 3] = v.w;
                }
                v8bf o0, o1;
#pragma unroll
                for (int j = 0; j < 16; j++) {
                    float nv = fmaf((vals[j] - mu) * rs, gam[kg + j], bet[kg + j]);
                    if (j < 8) o0[j] = (__bf16)nv; else o1[j - 8] = (__bf16)nv;
                }
                *(v8bf*)&Abuf[sr][seg * 16]     = o0;
                *(v8bf*)&Abuf[sr][seg * 16 + 8] = o1;
            }

            // current buffer's 2 loads (issued last iteration) complete in-order;
            // only the 2 just-issued may remain outstanding.
            tl_wait_async_le2();
            __syncthreads();

            // ---- 4 WMMAs per wave per k-chunk ----
            const __bf16* arow  = &Abuf[rowblk * 16 + lh][0];
            const __bf16* bcol0 = &Bbuf[buf][colgrp * 32 + lh][0];
            const __bf16* bcol1 = &Bbuf[buf][colgrp * 32 + 16 + lh][0];
#pragma unroll
            for (int ks = 0; ks < 64; ks += 32) {
                v16bf a   = loadA2(arow, ks + hb * 16, ks + hb * 16 + 8);
                v16bf b0  = load16c(bcol0 + ks + hb * 16);
                v16bf b1f = load16c(bcol1 + ks + hb * 16);
                acc0 = __builtin_amdgcn_wmma_f32_16x16x32_bf16(false, a, false, b0,
                                                               (short)0, acc0, false, false);
                acc1 = __builtin_amdgcn_wmma_f32_16x16x32_bf16(false, a, false, b1f,
                                                               (short)0, acc1, false, false);
            }
        }

        // ---- +b1, exact GELU, h tile -> LDS (C/D layout -> A layout round trip) ----
        __syncthreads();
        {
            const int nl0 = colgrp * 32 + lh;
            const float bia0 = b1[ct * 64 + nl0];
            const float bia1 = b1[ct * 64 + nl0 + 16];
#pragma unroll
            for (int j = 0; j < 8; j++) {
                const int m = rowblk * 16 + kb + j;   // C/D: lanes 16..31 hold rows +8
                Hbuf[m][nl0]      = (__bf16)gelu_exact(acc0[j] + bia0);
                Hbuf[m][nl0 + 16] = (__bf16)gelu_exact(acc1[j] + bia1);
            }
        }
        __syncthreads();

        // ---- GEMM2: logits[64x16] += GELU(h)[64x64] @ W2T-chunk, waves 0..3 ----
        if (colgrp == 0) {
            const __bf16* hrow = &Hbuf[rowblk * 16 + lh][0];
#pragma unroll
            for (int ks = 0; ks < 64; ks += 32) {
                v16bf a = loadA2(hrow, ks + hb * 16, ks + hb * 16 + 8);
                const __bf16* bp = W2T + (size_t)lh * C_ + ct * 64 + ks + hb * 16;
                v16bf bm = load16c(bp);
                lacc = __builtin_amdgcn_wmma_f32_16x16x32_bf16(false, a, false, bm,
                                                               (short)0, lacc, false, false);
            }
        }
    }

    // ---- write logits: cols 0..7 of the 16-wide C/D tile ----
    // (b2 is omitted on purpose: softmax over tokens is invariant to a constant
    //  per-(b,k) shift, and logits are only consumed by that softmax.)
    if (colgrp == 0 && lh < K_) {
        float* lp = logits + (m0 + rowblk * 16 + kb) * K_ + lh;
#pragma unroll
        for (int j = 0; j < 8; j++) lp[j * K_] = lacc[j];
    }
}

// ---------- kernel 2: softmax stats over tokens per (b,k) ----------
__global__ void tl_softmax_stats(const float* __restrict__ logits,
                                 float* __restrict__ mx, float* __restrict__ inv_sum) {
    const int bk = blockIdx.x;          // 0..255
    const int b = bk >> 3, k = bk & 7;
    const float* base = logits + (size_t)b * N_ * K_ + k;
    const int tid = threadIdx.x;
    __shared__ float red[8];

    float m = -3.402823466e+38f;
    for (int i = tid; i < N_; i += 256) m = fmaxf(m, base[(size_t)i * K_]);
#pragma unroll
    for (int off = 16; off; off >>= 1) m = fmaxf(m, __shfl_xor(m, off, 32));
    if ((tid & 31) == 0) red[tid >> 5] = m;
    __syncthreads();
    m = red[0];
#pragma unroll
    for (int i = 1; i < 8; i++) m = fmaxf(m, red[i]);
    __syncthreads();

    float s = 0.f;
    for (int i = tid; i < N_; i += 256) s += expf(base[(size_t)i * K_] - m);
#pragma unroll
    for (int off = 16; off; off >>= 1) s += __shfl_xor(s, off, 32);
    if ((tid & 31) == 0) red[tid >> 5] = s;
    __syncthreads();
    if (tid == 0) {
        float tot = 0.f;
#pragma unroll
        for (int i = 0; i < 8; i++) tot += red[i];
        mx[bk] = m;
        inv_sum[bk] = 1.f / tot;
    }
}

// ---------- kernel 3: weighted token pooling partials (deterministic 4-way) ----------
__global__ void __launch_bounds__(256)
tl_pool(const float* __restrict__ x, const float* __restrict__ logits,
        const float* __restrict__ mx, const float* __restrict__ inv_sum,
        float* __restrict__ partial) {
    const int cc  = blockIdx.x;         // 0..2  (256 channels each)
    const int b   = blockIdx.y;         // 0..31
    const int seg = blockIdx.z;         // 0..3  (1024 tokens each)
    const int tid = threadIdx.x;
    const int c = cc * 256 + tid;

    __shared__ float wbuf[128][K_];
    __shared__ float mxs[K_], invs[K_];
    if (tid < K_) { mxs[tid] = mx[b * K_ + tid]; invs[tid] = inv_sum[b * K_ + tid]; }

    float acc[K_];
#pragma unroll
    for (int k = 0; k < K_; k++) acc[k] = 0.f;

    for (int nb = seg * 1024; nb < seg * 1024 + 1024; nb += 128) {
        __syncthreads();
#pragma unroll
        for (int u = 0; u < 4; u++) {
            const int idx = tid * 4 + u;            // 0..1023 = 128 tokens x 8 slots
            const int nl = idx >> 3, kk = idx & 7;
            const float lg = logits[((size_t)b * N_ + nb + nl) * K_ + kk];
            wbuf[nl][kk] = expf(lg - mxs[kk]) * invs[kk];
        }
        __syncthreads();
        for (int nl = 0; nl < 128; nl++) {
            const float xv = x[((size_t)b * N_ + nb + nl) * C_ + c];
#pragma unroll
            for (int kk = 0; kk < K_; kk++) acc[kk] = fmaf(wbuf[nl][kk], xv, acc[kk]);
        }
    }

    float* pb = partial + (size_t)seg * (B_ * K_ * C_) + (size_t)b * K_ * C_ + c;
#pragma unroll
    for (int kk = 0; kk < K_; kk++) pb[(size_t)kk * C_] = acc[kk];
}

// ---------- kernel 4: sum the 4 deterministic partials ----------
__global__ void tl_reduce(const float* __restrict__ partial, float* __restrict__ out) {
    const int i = blockIdx.x * 256 + threadIdx.x;
    const int tot = B_ * K_ * C_;
    if (i < tot)
        out[i] = partial[i] + partial[tot + i] + partial[2 * tot + i] + partial[3 * tot + i];
}

// ---------- launch ----------
extern "C" void kernel_launch(void* const* d_in, const int* in_sizes, int n_in,
                              void* d_out, int out_size, void* d_ws, size_t ws_size,
                              hipStream_t stream) {
    const float* x   = (const float*)d_in[0];
    const float* gam = (const float*)d_in[1];
    const float* bet = (const float*)d_in[2];
    const float* W1  = (const float*)d_in[3];
    const float* b1  = (const float*)d_in[4];
    const float* W2  = (const float*)d_in[5];
    float* out = (float*)d_out;

    char* ws = (char*)d_ws;
    __bf16* W1T    = (__bf16*)ws;                                  // 768*768*2   = 1179648 B
    __bf16* W2T    = (__bf16*)(ws + 1179648);                      // 16*768*2    = 24576 B
    float*  logits = (float*)(ws + 1179648 + 24576);               // 131072*8*4  = 4194304 B
    float*  mx     = (float*)(ws + 1179648 + 24576 + 4194304);     // 256 * 4
    float*  invs   = mx + 256;                                     // 256 * 4
    float*  partial = invs + 256;                                  // 4*196608*4  = 3145728 B

    tl_prep<<<(C_ * C_ + 255) / 256, 256, 0, stream>>>(W1, W2, W1T, W2T);
    tl_fused<<<MTOT / MT, 256, 0, stream>>>(x, gam, bet, W1T, b1, W2T, logits);
    tl_softmax_stats<<<B_ * K_, 256, 0, stream>>>(logits, mx, invs);
    tl_pool<<<dim3(C_ / 256, B_, 4), 256, 0, stream>>>(x, logits, mx, invs, partial);
    tl_reduce<<<(B_ * K_ * C_ + 255) / 256, 256, 0, stream>>>(partial, out);
}
